// HCF_2851858284699
// MI455X (gfx1250) — matrix operand order repaired and checked by reading, not compiled, not run on hardware
//
#include <hip/hip_runtime.h>
#include <hip/hip_bf16.h>
#include <math.h>

// ---------------------------------------------------------------------------
// CDNA5 (gfx1250) implementation: dual cross-modal attention fusion.
// All heavy GEMMs use v_wmma_f32_16x16x32_bf16 (wave32).
// Flash-style attention: per-row (max, 1/Z) stats precomputed with a
// branch-free online softmax pass, so the [B,N,N] probability matrices are
// never materialized.
// ---------------------------------------------------------------------------

typedef __bf16 bf16_t;
typedef __attribute__((ext_vector_type(16))) __bf16 v16bf;
typedef __attribute__((ext_vector_type(8)))  __bf16 v8bf;
typedef __attribute__((ext_vector_type(8)))  float  v8f;
typedef __attribute__((ext_vector_type(4)))  float  v4f;

#define DEV __device__ __forceinline__

static constexpr int   B_  = 8;
static constexpr int   C_  = 256;
static constexpr int   H_  = 48;
static constexpr int   W_  = 48;
static constexpr int   N_  = H_ * W_;     // 2304
static constexpr int   CR_ = 32;
static constexpr float BN_SCALE_ = 0.99999500003749975f;   // 1/sqrt(1+1e-5)
static constexpr float QK_SCALE_ = 0.17677669529663687f;   // 32^-0.5

DEV float sigm(float x) { return 1.0f / (1.0f + __expf(-x)); }

DEV v8f wmma_bf16(v16bf a, v16bf b, v8f c) {
    return __builtin_amdgcn_wmma_f32_16x16x32_bf16(
        /*neg_a=*/false, a, /*neg_b=*/false, b,
        /*c_mod=*/(short)0, c, /*reuse_a=*/false, /*reuse_b=*/false);
}

// A fragment (16 rows x 32 K, bf16, row-major source, ld in elements).
// Lane (l = lane&15, h = lane>>4): row = row0 + l; element i holds K=16h+i
// -> two contiguous 8-element (16B) chunks.
DEV v16bf load_a_frag_bf(const bf16_t* __restrict__ base, int ld,
                         int row0, int k0, int lane) {
    int l = lane & 15, h = lane >> 4;
    const bf16_t* r = base + (size_t)(row0 + l) * ld + k0 + 16 * h;
    v8bf c0 = *(const v8bf*)r;
    v8bf c1 = *(const v8bf*)(r + 8);
    v16bf a;
#pragma unroll
    for (int i = 0; i < 8; ++i) { a[i] = c0[i]; a[8 + i] = c1[i]; }
    return a;
}

// Same A layout but converting from fp32 with a premultiplier (for weights).
DEV v16bf load_a_frag_f32(const float* __restrict__ base, int ld,
                          int row0, int k0, int lane, float mul) {
    int l = lane & 15, h = lane >> 4;
    const float* r = base + (size_t)(row0 + l) * ld + k0 + 16 * h;
    v16bf a;
#pragma unroll
    for (int i = 0; i < 8; ++i) {
        a[i]     = (bf16_t)(r[i] * mul);
        a[8 + i] = (bf16_t)(r[8 + i] * mul);
    }
    return a;
}

// B fragment (32 K x 16 cols, row-major source): col = col0 + (lane&15),
// VGPR idx holds K = 16h + idx (stride-ld gathers).
DEV v16bf load_b_frag_bf(const bf16_t* __restrict__ base, int ld,
                         int k0, int col0, int lane) {
    int l = lane & 15, h = lane >> 4;
    const bf16_t* p = base + (size_t)(k0 + 16 * h) * ld + col0 + l;
    v16bf b;
#pragma unroll
    for (int i = 0; i < 16; ++i) b[i] = p[(size_t)i * ld];
    return b;
}

DEV v16bf load_b_frag_f32(const float* __restrict__ base, int ld,
                          int k0, int col0, int lane) {
    int l = lane & 15, h = lane >> 4;
    const float* p = base + (size_t)(k0 + 16 * h) * ld + col0 + l;
    v16bf b;
#pragma unroll
    for (int i = 0; i < 16; ++i) b[i] = (bf16_t)p[(size_t)i * ld];
    return b;
}

// ---------------------------------------------------------------------------
// Projection GEMM: Y[o,n] = sum_k W[o,k] * X[b,k,n] + bias[o], output bf16.
// Compile-time K / COUT / TRANS: fully unrolled WMMA chain, branch-free
// stores.  TRANS=1 stores Y as [B,N,COUT] with one packed 16B store per lane
// (the 8 accumulator rows are contiguous in that layout); premul folds the
// softmax scale into Q before bf16 quantization.
// grid: (N/64, COUT/16, B), block 128 (4 waves, one 16x16 tile each).
// ---------------------------------------------------------------------------
template <int K, int COUT, int TRANS>
__global__ void __launch_bounds__(128)
k_proj(const float* __restrict__ X, const float* __restrict__ Wm,
       const float* __restrict__ bias, bf16_t* __restrict__ Y, float premul) {
    int lane = threadIdx.x & 31, w = threadIdx.x >> 5;
    int l = lane & 15, h = lane >> 4;
    int col0 = (blockIdx.x * 4 + w) * 16;
    int row0 = blockIdx.y * 16;
    int b = blockIdx.z;
    const float* Xb = X + (size_t)b * K * N_;
    v8f acc = {};
#pragma unroll
    for (int k0 = 0; k0 < K; k0 += 32) {
        v16bf a  = load_a_frag_f32(Wm, K, row0, k0, lane, premul);
        v16bf bb = load_b_frag_f32(Xb, N_, k0, col0, lane);
        acc = wmma_bf16(a, bb, acc);
    }
    if (TRANS) {
        v8bf o;
#pragma unroll
        for (int g = 0; g < 8; ++g)
            o[g] = (bf16_t)(acc[g] + bias[row0 + 8 * h + g] * premul);
        *(v8bf*)(Y + ((size_t)b * N_ + col0 + l) * COUT + row0 + 8 * h) = o;
    } else {
#pragma unroll
        for (int g = 0; g < 8; ++g) {
            int row = row0 + g + 8 * h;
            Y[((size_t)b * COUT + row) * (size_t)N_ + col0 + l] =
                (bf16_t)(acc[g] + bias[row] * premul);
        }
    }
}

// ---------------------------------------------------------------------------
// Pass 1: per-row softmax stats for S = qT (N x 32) * K (32 x N).
// Branch-free online (max,sumexp) per lane across the wave's m-tile stripe
// (18 unrolled WMMA trips): M=max(rm,x); rs=rs*exp(rm-M)+exp(x-M).  Then
// shfl-xor reduction over the 16 columns and an LDS reduction over 8 waves.
// grid: (N/16, B), block 256.
// ---------------------------------------------------------------------------
__global__ void __launch_bounds__(256)
k_rowstats(const bf16_t* __restrict__ qT, const bf16_t* __restrict__ Kmat,
           float* __restrict__ rowmax, float* __restrict__ invz) {
    int tid = threadIdx.x, lane = tid & 31, w = tid >> 5;
    int l = lane & 15;
    int h = lane >> 4;
    int n0 = blockIdx.x * 16, b = blockIdx.y;
    const bf16_t* qb = qT   + (size_t)b * N_ * CR_;
    const bf16_t* kb = Kmat + (size_t)b * CR_ * N_;
    v16bf a = load_a_frag_bf(qb, CR_, n0, 0, lane);

    float rm[8], rs[8];
#pragma unroll
    for (int g = 0; g < 8; ++g) { rm[g] = -1e30f; rs[g] = 0.0f; }

#pragma unroll
    for (int it = 0; it < (N_ / 16) / 8; ++it) {   // 18 trips
        int mt = w + it * 8;
        v16bf bb = load_b_frag_bf(kb, N_, 0, mt * 16, lane);
        v8f z = {};
        v8f s = wmma_bf16(a, bb, z);
#pragma unroll
        for (int g = 0; g < 8; ++g) {
            float x = s[g];
            float M = fmaxf(rm[g], x);                       // branch-free
            rs[g] = rs[g] * __expf(rm[g] - M) + __expf(x - M);
            rm[g] = M;
        }
    }
    // combine across the 16 columns of each half-wave
#pragma unroll
    for (int mask = 1; mask < 16; mask <<= 1) {
#pragma unroll
        for (int g = 0; g < 8; ++g) {
            float om = __shfl_xor(rm[g], mask, 32);
            float os = __shfl_xor(rs[g], mask, 32);
            float M  = fmaxf(rm[g], om);
            rs[g] = rs[g] * __expf(rm[g] - M) + os * __expf(om - M);
            rm[g] = M;
        }
    }
    __shared__ float smax[8][16], ssum[8][16];
    if (l == 0) {
#pragma unroll
        for (int g = 0; g < 8; ++g) {
            smax[w][8 * h + g] = rm[g];
            ssum[w][8 * h + g] = rs[g];
        }
    }
    __syncthreads();
    if (tid < 16) {
        float M = smax[0][tid], S = ssum[0][tid];
        for (int ww = 1; ww < 8; ++ww) {
            float om = smax[ww][tid], os = ssum[ww][tid];
            float M2 = fmaxf(M, om);
            S = S * __expf(M - M2) + os * __expf(om - M2);
            M = M2;
        }
        rowmax[(size_t)b * N_ + n0 + tid] = M;
        invz  [(size_t)b * N_ + n0 + tid] = 1.0f / S;
    }
}

// ---------------------------------------------------------------------------
// Pass 2: O[c,m] = sum_n V[c,n] * exp(S[n,m]-rowmax[n])*invz[n].
// Block covers a [256 c x 128 m] output tile with 16 waves (one 16-row c
// block each). Per 32-wide n block: waves 0-7 regenerate the score tile via
// WMMA, exponentiate, and stage bf16 P in LDS *transposed* (P2[m][n], row
// stride 40 elements = 80B so the 16h offset is 16B-aligned): a lane's 16
// B-fragment elements become contiguous -> two ds_load_b128 instead of 16
// ds_load_u16.  Then every wave runs one 16x16x32 WMMA per m-subtile.
// grid: (N/128, B), block 512.
// ---------------------------------------------------------------------------
__global__ void __launch_bounds__(512)
k_attn(const bf16_t* __restrict__ qT, const bf16_t* __restrict__ Kmat,
       const bf16_t* __restrict__ V, const float* __restrict__ rowmax,
       const float* __restrict__ invz, bf16_t* __restrict__ att) {
    int tid = threadIdx.x, lane = tid & 31, w = tid >> 5;
    int l = lane & 15, h = lane >> 4;
    int m0 = blockIdx.x * 128, b = blockIdx.y;
    const bf16_t* qb  = qT   + (size_t)b * N_ * CR_;
    const bf16_t* kb  = Kmat + (size_t)b * CR_ * N_;
    const bf16_t* vb  = V    + (size_t)b * C_ * N_;
    const float*  rmb = rowmax + (size_t)b * N_;
    const float*  izb = invz   + (size_t)b * N_;

    __shared__ bf16_t P2[128][40];  // [m_local][n_local], 80B row stride

    int c0 = w * 16;
    // K fragment for the score GEMM is constant over the n loop.
    v16bf bs = load_b_frag_bf(kb, N_, 0, m0 + 16 * (w & 7), lane);

    v8f zero = {};
    v8f acc[8];
#pragma unroll
    for (int j = 0; j < 8; ++j) acc[j] = zero;

    for (int nb = 0; nb < N_; nb += 32) {
        if (w < 8) {
            v16bf a0 = load_a_frag_bf(qb, CR_, nb,      0, lane);
            v16bf a1 = load_a_frag_bf(qb, CR_, nb + 16, 0, lane);
            v8f s0 = wmma_bf16(a0, bs, zero);
            v8f s1 = wmma_bf16(a1, bs, zero);
            // vectorized per-row stats (contiguous in g): 4x float4 each
            const v4f* prm = (const v4f*)(rmb + nb + 8 * h);
            const v4f* piz = (const v4f*)(izb + nb + 8 * h);
            v4f ra = prm[0], rb2 = prm[1], rc = prm[4], rd = prm[5];
            v4f ia = piz[0], ib2 = piz[1], ic = piz[4], id2 = piz[5];
            int ml = 16 * w + l;           // m_local of this lane's column
#pragma unroll
            for (int g = 0; g < 4; ++g) {
                P2[ml][g + 8 * h]      = (bf16_t)(__expf(s0[g] - ra[g]) * ia[g]);
                P2[ml][16 + g + 8 * h] = (bf16_t)(__expf(s1[g] - rc[g]) * ic[g]);
            }
#pragma unroll
            for (int g = 0; g < 4; ++g) {
                P2[ml][4 + g + 8 * h]  = (bf16_t)(__expf(s0[4 + g] - rb2[g]) * ib2[g]);
                P2[ml][20 + g + 8 * h] = (bf16_t)(__expf(s1[4 + g] - rd[g]) * id2[g]);
            }
        }
        __syncthreads();
        v16bf av = load_a_frag_bf(vb, N_, c0, nb, lane);
#pragma unroll
        for (int j = 0; j < 8; ++j) {
            // B fragment: K = 16h+i at column 16j+l -> contiguous in LDS.
            const v8bf* pp = (const v8bf*)&P2[16 * j + l][16 * h];
            v8bf lo = pp[0], hi = pp[1];
            v16bf bp;
#pragma unroll
            for (int i = 0; i < 8; ++i) { bp[i] = lo[i]; bp[8 + i] = hi[i]; }
            acc[j] = wmma_bf16(av, bp, acc[j]);
        }
        __syncthreads();
    }
#pragma unroll
    for (int j = 0; j < 8; ++j)
#pragma unroll
        for (int g = 0; g < 8; ++g)
            att[((size_t)b * C_ + c0 + g + 8 * h) * N_ + m0 + 16 * j + l] =
                (bf16_t)acc[j][g];
}

// ---------------------------------------------------------------------------
// Gate GEMM (K=512 over [rgb_att; chm_att], fully unrolled -> 16 WMMAs) +
// sigmoid + fusion: fused = r*g + d*(1-g) + r*d.
// grid: (N/64, C/16, B), block 128.
// ---------------------------------------------------------------------------
__global__ void __launch_bounds__(128)
k_gate_fuse(const float* __restrict__ gw, const float* __restrict__ gb,
            const bf16_t* __restrict__ att1, const bf16_t* __restrict__ att2,
            float* __restrict__ fused) {
    int lane = threadIdx.x & 31, w = threadIdx.x >> 5;
    int l = lane & 15, h = lane >> 4;
    int col0 = (blockIdx.x * 4 + w) * 16;
    int row0 = blockIdx.y * 16;
    int b = blockIdx.z;
    const bf16_t* a1 = att1 + (size_t)b * C_ * N_;
    const bf16_t* a2 = att2 + (size_t)b * C_ * N_;
    v8f acc = {};
#pragma unroll
    for (int k0 = 0; k0 < 2 * C_; k0 += 32) {
        v16bf a = load_a_frag_f32(gw, 2 * C_, row0, k0, lane, 1.0f);
        const bf16_t* src = (k0 < C_) ? (a1 + (size_t)k0 * N_)
                                      : (a2 + (size_t)(k0 - C_) * N_);
        v16bf bb = load_b_frag_bf(src, N_, 0, col0, lane);
        acc = wmma_bf16(a, bb, acc);
    }
#pragma unroll
    for (int g = 0; g < 8; ++g) {
        int c = row0 + g + 8 * h;
        size_t idx = ((size_t)b * C_ + c) * N_ + col0 + l;
        float gv = sigm(acc[g] + gb[c]);
        float r  = (float)att1[idx];
        float d  = (float)att2[idx];
        fused[idx] = r * gv + d * (1.0f - gv) + r * d;
    }
}

// ---------------------------------------------------------------------------
// MLP head: 256 -> 24 (BN+ReLU) -> 1 (BN+sigmoid) -> sigmoid.  One thread per
// pixel; mlp1 weights staged in LDS (24 KB).  grid: B*N/256, block 256.
// ---------------------------------------------------------------------------
__global__ void __launch_bounds__(256)
k_mlp(const float* __restrict__ fused, const float* __restrict__ w1,
      const float* __restrict__ g1, const float* __restrict__ b1,
      const float* __restrict__ w2, const float* __restrict__ g2,
      const float* __restrict__ b2, float* __restrict__ wgt) {
    __shared__ float W1s[24 * 256];
    for (int i = threadIdx.x; i < 24 * 256; i += blockDim.x) W1s[i] = w1[i];
    __syncthreads();
    int idx = blockIdx.x * blockDim.x + threadIdx.x;
    int b = idx / N_, n = idx % N_;
    const float* fb = fused + (size_t)b * C_ * N_ + n;
    float acc[24];
#pragma unroll
    for (int j = 0; j < 24; ++j) acc[j] = 0.0f;
    for (int c = 0; c < C_; ++c) {
        float f = fb[(size_t)c * N_];
#pragma unroll
        for (int j = 0; j < 24; ++j) acc[j] += W1s[j * 256 + c] * f;
    }
    float s2 = 0.0f;
#pragma unroll
    for (int j = 0; j < 24; ++j) {
        float hj = fmaxf(acc[j] * (BN_SCALE_ * g1[j]) + b1[j], 0.0f);
        s2 += w2[j] * hj;
    }
    float y = sigm(s2 * (BN_SCALE_ * g2[0]) + b2[0]);
    wgt[idx] = sigm(y);   // double sigmoid, as in reference
}

// --------------------------- CBAM helper kernels ---------------------------
__global__ void k_avgmax(const float* __restrict__ x, float* __restrict__ am) {
    int idx = blockIdx.x * blockDim.x + threadIdx.x;   // B*N
    int b = idx / N_, n = idx % N_;
    const float* xb = x + (size_t)b * C_ * N_ + n;
    float s = 0.0f, m = -1e30f;
    for (int c = 0; c < C_; ++c) { float v = xb[(size_t)c * N_]; s += v; m = fmaxf(m, v); }
    am[((size_t)b * 2 + 0) * N_ + n] = s * (1.0f / C_);
    am[((size_t)b * 2 + 1) * N_ + n] = m;
}

__global__ void k_sconv(const float* __restrict__ am, const float* __restrict__ wc,
                        const float* __restrict__ bc, float* __restrict__ sa) {
    int idx = blockIdx.x * blockDim.x + threadIdx.x;   // B*N
    int b = idx / N_, n = idx % N_;
    int y = n / W_, x = n % W_;
    const float* a0 = am + (size_t)b * 2 * N_;
    float s = bc[0];
#pragma unroll
    for (int ch = 0; ch < 2; ++ch)
        for (int dy = 0; dy < 7; ++dy) {
            int yy = y + dy - 3;
            if (yy < 0 || yy >= H_) continue;
            for (int dx = 0; dx < 7; ++dx) {
                int xx = x + dx - 3;
                if (xx < 0 || xx >= W_) continue;
                s += wc[ch * 49 + dy * 7 + dx] * a0[(size_t)ch * N_ + yy * W_ + xx];
            }
        }
    sa[idx] = sigm(s);
}

__global__ void k_chanstat(const float* __restrict__ x, float* __restrict__ cavg,
                           float* __restrict__ cmax) {
    int bc = blockIdx.x;                   // B*C
    const float* xb = x + (size_t)bc * N_;
    float s = 0.0f, m = -1e30f;
    for (int i = threadIdx.x; i < N_; i += blockDim.x) {
        float v = xb[i]; s += v; m = fmaxf(m, v);
    }
    __shared__ float ss[128], sm[128];
    ss[threadIdx.x] = s; sm[threadIdx.x] = m;
    __syncthreads();
    for (int st = 64; st > 0; st >>= 1) {
        if (threadIdx.x < st) {
            ss[threadIdx.x] += ss[threadIdx.x + st];
            sm[threadIdx.x] = fmaxf(sm[threadIdx.x], sm[threadIdx.x + st]);
        }
        __syncthreads();
    }
    if (threadIdx.x == 0) { cavg[bc] = ss[0] * (1.0f / N_); cmax[bc] = sm[0]; }
}

__global__ void k_chanfc(const float* __restrict__ cavg, const float* __restrict__ cmax,
                         const float* __restrict__ w1, const float* __restrict__ b1,
                         const float* __restrict__ w2, const float* __restrict__ b2,
                         float* __restrict__ ca) {
    int b = blockIdx.x;
    __shared__ float va[256], vm[256], hid[32];
    int t = threadIdx.x;
    va[t] = cavg[b * 256 + t];
    vm[t] = cmax[b * 256 + t];
    __syncthreads();
    if (t < 32) {
        int j = t & 15;
        const float* v = (t < 16) ? va : vm;
        float s = b1[j];
        for (int c = 0; c < 256; ++c) s += w1[j * 256 + c] * v[c];
        hid[t] = fmaxf(s, 0.0f);
    }
    __syncthreads();
    float s = 2.0f * b2[t];
#pragma unroll
    for (int j = 0; j < 16; ++j) s += w2[t * 16 + j] * (hid[j] + hid[16 + j]);
    ca[b * 256 + t] = sigm(s);
}

// final = rgb*chm_sa*chm_ca + chm*rgb_sa*rgb_ca*weight
__global__ void k_final(const float* __restrict__ rgb, const float* __restrict__ chm,
                        const float* __restrict__ sa_r, const float* __restrict__ sa_c,
                        const float* __restrict__ ca_r, const float* __restrict__ ca_c,
                        const float* __restrict__ wgt, float* __restrict__ out) {
    size_t idx = (size_t)blockIdx.x * blockDim.x + threadIdx.x;  // B*C*N
    int n = (int)(idx % N_);
    size_t bc = idx / N_;
    int b = (int)(bc / C_);
    float ar = rgb[idx] * sa_c[(size_t)b * N_ + n] * ca_c[bc];
    float ac = chm[idx] * sa_r[(size_t)b * N_ + n] * ca_r[bc];
    out[idx] = ar + ac * wgt[(size_t)b * N_ + n];
}

// ---------------------------------------------------------------------------
extern "C" void kernel_launch(void* const* d_in, const int* in_sizes, int n_in,
                              void* d_out, int out_size, void* d_ws, size_t ws_size,
                              hipStream_t stream) {
    (void)in_sizes; (void)n_in; (void)out_size; (void)ws_size;
    const float* rgb       = (const float*)d_in[0];
    const float* chm       = (const float*)d_in[1];
    const float* rgb_sa_w  = (const float*)d_in[2];
    const float* rgb_sa_b  = (const float*)d_in[3];
    const float* chm_sa_w  = (const float*)d_in[4];
    const float* chm_sa_b  = (const float*)d_in[5];
    const float* rgb_fc1_w = (const float*)d_in[6];
    const float* rgb_fc1_b = (const float*)d_in[7];
    const float* rgb_fc2_w = (const float*)d_in[8];
    const float* rgb_fc2_b = (const float*)d_in[9];
    const float* chm_fc1_w = (const float*)d_in[10];
    const float* chm_fc1_b = (const float*)d_in[11];
    const float* chm_fc2_w = (const float*)d_in[12];
    const float* chm_fc2_b = (const float*)d_in[13];
    const float* rq_w = (const float*)d_in[14];
    const float* rq_b = (const float*)d_in[15];
    const float* rk_w = (const float*)d_in[16];
    const float* rk_b = (const float*)d_in[17];
    const float* rv_w = (const float*)d_in[18];
    const float* rv_b = (const float*)d_in[19];
    const float* dq_w = (const float*)d_in[20];
    const float* dq_b = (const float*)d_in[21];
    const float* dk_w = (const float*)d_in[22];
    const float* dk_b = (const float*)d_in[23];
    const float* dv_w = (const float*)d_in[24];
    const float* dv_b = (const float*)d_in[25];
    const float* gate_w = (const float*)d_in[26];
    const float* gate_b = (const float*)d_in[27];
    const float* mlp1_w = (const float*)d_in[28];
    const float* mlp1_g = (const float*)d_in[29];
    const float* mlp1_b = (const float*)d_in[30];
    const float* mlp2_w = (const float*)d_in[31];
    const float* mlp2_g = (const float*)d_in[32];
    const float* mlp2_b = (const float*)d_in[33];

    char* ws = (char*)d_ws;
    size_t off = 0;
    auto alloc = [&](size_t bytes) -> void* {
        void* p = ws + off;
        off += (bytes + 255) & ~(size_t)255;
        return p;
    };

    bf16_t* rqT  = (bf16_t*)alloc((size_t)B_ * N_ * CR_ * 2);
    bf16_t* dqT  = (bf16_t*)alloc((size_t)B_ * N_ * CR_ * 2);
    bf16_t* rkM  = (bf16_t*)alloc((size_t)B_ * CR_ * N_ * 2);
    bf16_t* dkM  = (bf16_t*)alloc((size_t)B_ * CR_ * N_ * 2);
    bf16_t* rvM  = (bf16_t*)alloc((size_t)B_ * C_ * N_ * 2);
    bf16_t* dvM  = (bf16_t*)alloc((size_t)B_ * C_ * N_ * 2);
    float*  rm1  = (float*) alloc((size_t)B_ * N_ * 4);
    float*  iz1  = (float*) alloc((size_t)B_ * N_ * 4);
    float*  rm2  = (float*) alloc((size_t)B_ * N_ * 4);
    float*  iz2  = (float*) alloc((size_t)B_ * N_ * 4);
    bf16_t* att1 = (bf16_t*)alloc((size_t)B_ * C_ * N_ * 2);
    bf16_t* att2 = (bf16_t*)alloc((size_t)B_ * C_ * N_ * 2);
    float*  fus  = (float*) alloc((size_t)B_ * C_ * N_ * 4);
    float*  am_r = (float*) alloc((size_t)B_ * 2 * N_ * 4);
    float*  am_c = (float*) alloc((size_t)B_ * 2 * N_ * 4);
    float*  sa_r = (float*) alloc((size_t)B_ * N_ * 4);
    float*  sa_c = (float*) alloc((size_t)B_ * N_ * 4);
    float*  cavg_r = (float*)alloc((size_t)B_ * C_ * 4);
    float*  cmax_r = (float*)alloc((size_t)B_ * C_ * 4);
    float*  cavg_c = (float*)alloc((size_t)B_ * C_ * 4);
    float*  cmax_c = (float*)alloc((size_t)B_ * C_ * 4);
    float*  ca_r = (float*) alloc((size_t)B_ * C_ * 4);
    float*  ca_c = (float*) alloc((size_t)B_ * C_ * 4);
    float*  wgt  = (float*) alloc((size_t)B_ * N_ * 4);

    // CBAM branch (cheap, VALU)
    k_avgmax<<<B_ * N_ / 256, 256, 0, stream>>>(rgb, am_r);
    k_avgmax<<<B_ * N_ / 256, 256, 0, stream>>>(chm, am_c);
    k_sconv<<<B_ * N_ / 256, 256, 0, stream>>>(am_r, rgb_sa_w, rgb_sa_b, sa_r);
    k_sconv<<<B_ * N_ / 256, 256, 0, stream>>>(am_c, chm_sa_w, chm_sa_b, sa_c);
    k_chanstat<<<B_ * C_, 128, 0, stream>>>(rgb, cavg_r, cmax_r);
    k_chanstat<<<B_ * C_, 128, 0, stream>>>(chm, cavg_c, cmax_c);
    k_chanfc<<<B_, 256, 0, stream>>>(cavg_r, cmax_r, rgb_fc1_w, rgb_fc1_b,
                                     rgb_fc2_w, rgb_fc2_b, ca_r);
    k_chanfc<<<B_, 256, 0, stream>>>(cavg_c, cmax_c, chm_fc1_w, chm_fc1_b,
                                     chm_fc2_w, chm_fc2_b, ca_c);

    // Projections (WMMA). Softmax scale folded into Q before bf16 rounding.
    dim3 gq(N_ / 64, CR_ / 16, B_);
    dim3 gv(N_ / 64, C_ / 16, B_);
    k_proj<C_, CR_, 1><<<gq, 128, 0, stream>>>(rgb, rq_w, rq_b, rqT, QK_SCALE_);
    k_proj<C_, CR_, 1><<<gq, 128, 0, stream>>>(chm, dq_w, dq_b, dqT, QK_SCALE_);
    k_proj<C_, CR_, 0><<<gq, 128, 0, stream>>>(rgb, rk_w, rk_b, rkM, 1.0f);
    k_proj<C_, CR_, 0><<<gq, 128, 0, stream>>>(chm, dk_w, dk_b, dkM, 1.0f);
    k_proj<C_, C_, 0><<<gv, 128, 0, stream>>>(rgb, rv_w, rv_b, rvM, 1.0f);
    k_proj<C_, C_, 0><<<gv, 128, 0, stream>>>(chm, dv_w, dv_b, dvM, 1.0f);

    // Flash-style cross attention (WMMA score regen + WMMA V*P accumulate)
    dim3 gs(N_ / 16, B_);
    k_rowstats<<<gs, 256, 0, stream>>>(rqT, dkM, rm1, iz1);
    k_rowstats<<<gs, 256, 0, stream>>>(dqT, rkM, rm2, iz2);
    dim3 ga(N_ / 128, B_);
    k_attn<<<ga, 512, 0, stream>>>(rqT, dkM, rvM, rm1, iz1, att1);
    k_attn<<<ga, 512, 0, stream>>>(dqT, rkM, dvM, rm2, iz2, att2);

    // Gate GEMM (WMMA, K=512) + fusion, then MLP weight, then final blend.
    k_gate_fuse<<<dim3(N_ / 64, C_ / 16, B_), 128, 0, stream>>>(gate_w, gate_b,
                                                                att1, att2, fus);
    k_mlp<<<B_ * N_ / 256, 256, 0, stream>>>(fus, mlp1_w, mlp1_g, mlp1_b,
                                             mlp2_w, mlp2_g, mlp2_b, wgt);
    k_final<<<(int)(((size_t)B_ * C_ * N_) / 256), 256, 0, stream>>>(
        rgb, chm, sa_r, sa_c, ca_r, ca_c, wgt, (float*)d_out);
}
